// AugmentedGRU_47012712022103
// MI455X (gfx1250) — matrix operand order repaired and checked by reading, not compiled, not run on hardware
//
#include <hip/hip_runtime.h>
#include <hip/hip_bf16.h>

// ---------------------------------------------------------------------------
// AugmentedGRU for MI455X (gfx1250), wave32 + WMMA bf16.
//  Pass 0: x (f32) -> x_bf (bf16), one memory-bound sweep (~8us at 23.3TB/s).
//  Phase A: gx = x @ w_ih^T + b_ih  -- 64x64 wave tiles, pure bf16 b128 loads,
//           v_wmma_f32_16x16x32_bf16, 1 b128 load per WMMA.
//  Phase B: persistent 32-WG scan; w_hh slice LDS-resident, h ping-pong bf16,
//           one 32-arrival grid barrier per timestep.
//  Time processed in 16 chunks of 64 steps; A/B alternate on the stream so a
//  single 25MB gx buffer stays L2-resident (192MB L2).
// ---------------------------------------------------------------------------

typedef __bf16 bf16_t;
typedef __attribute__((ext_vector_type(16))) __bf16 v16bf;
typedef __attribute__((ext_vector_type(8)))  __bf16 v8bf;
typedef __attribute__((ext_vector_type(8)))  float  v8f;
typedef __attribute__((ext_vector_type(4)))  float  v4f;

#define T_DIM 1024
#define B_DIM 64
#define I_DIM 512
#define H_DIM 512
#define G3    (3 * H_DIM)          // 1536
#define TC    64                   // time-chunk
#define NCHUNK (T_DIM / TC)        // 16
#define MC    (TC * B_DIM)         // 4096 rows per chunk
#define KT    (I_DIM / 32)         // 16 k-tiles (K blocks of 32)
#define NT    (G3 / 16)            // 96 n-tiles
#define NWG_B (H_DIM / 16)         // 32 scan workgroups

// ---- helpers --------------------------------------------------------------

__device__ __forceinline__ bf16_t f2bf(float f) {
    union { float f; unsigned u; } in; in.f = f;
    unsigned u = in.u + (0x7FFFu + ((in.u >> 16) & 1u));   // round-to-nearest-even
    union { unsigned short s; bf16_t b; } out;
    out.s = (unsigned short)(u >> 16);
    return out.b;
}

__device__ __forceinline__ v8f zero8() {
    v8f c;
#pragma unroll
    for (int v = 0; v < 8; ++v) c[v] = 0.f;
    return c;
}

__device__ __forceinline__ v8f wmma_bf16(v16bf a, v16bf b, v8f c) {
    return __builtin_amdgcn_wmma_f32_16x16x32_bf16(false, a, false, b,
                                                   (short)0, c, false, false);
}

// A-fragment (16x32 bf16) from a row-major bf16 row. Per ISA 7.12.2:
// lane<16 holds K = [k0..k0+7] and [k0+16..k0+23], k0 = 32*kt (+8 for lanes>=16).
__device__ __forceinline__ v16bf load_a_bf(const bf16_t* __restrict__ row, int k0) {
    v8bf lo = *(const v8bf*)(row + k0);
    v8bf hi = *(const v8bf*)(row + k0 + 16);
    v16bf a;
#pragma unroll
    for (int e = 0; e < 8; ++e) { a[e] = lo[e]; a[e + 8] = hi[e]; }
    return a;
}

// B-fragment from fragment-major packed weights: block (kt,nt) = 32 lanes x 16 bf16.
__device__ __forceinline__ v16bf load_b_pk(const bf16_t* __restrict__ pk,
                                           int kt, int nt, int lane) {
    const bf16_t* p = pk + ((size_t)(kt * NT + nt) * 32 + lane) * 16;
    v8bf lo = *(const v8bf*)p;
    v8bf hi = *(const v8bf*)(p + 8);
    v16bf b;
#pragma unroll
    for (int e = 0; e < 8; ++e) { b[e] = lo[e]; b[e + 8] = hi[e]; }
    return b;
}

__device__ __forceinline__ v16bf load_b_lds(const bf16_t* lds, int gi, int kt, int lane) {
    const bf16_t* p = lds + ((gi * KT + kt) * 512) + lane * 16;
    v8bf lo = *(const v8bf*)p;
    v8bf hi = *(const v8bf*)(p + 8);
    v16bf b;
#pragma unroll
    for (int e = 0; e < 8; ++e) { b[e] = lo[e]; b[e + 8] = hi[e]; }
    return b;
}

__device__ __forceinline__ float sigmoid_f(float x) {
    return 1.f / (1.f + __expf(-x));
}
__device__ __forceinline__ float tanh_f(float x) {
    return 2.f / (1.f + __expf(-2.f * x)) - 1.f;   // v_exp_f32 trans pipe
}

// ---- kernel 0: zero state + barriers --------------------------------------

__global__ void init_state_kernel(float* h_f32, bf16_t* h_bf2, int* bars) {
    int i = blockIdx.x * blockDim.x + threadIdx.x;
    if (i < B_DIM * H_DIM) h_f32[i] = 0.f;
    if (i < 2 * B_DIM * H_DIM) h_bf2[i] = f2bf(0.f);
    if (i < 64) bars[i] = 0;
}

// ---- kernel 1: x (f32) -> x_bf (bf16), 8 elements per thread ---------------

__global__ void cvt_x_kernel(const float* __restrict__ x,
                             bf16_t* __restrict__ x_bf, int n8) {
    int i = blockIdx.x * blockDim.x + threadIdx.x;
    if (i >= n8) return;
    const float* p = x + (size_t)i * 8;
    v4f f0 = *(const v4f*)p;
    v4f f1 = *(const v4f*)(p + 4);
    v8bf o;
#pragma unroll
    for (int e = 0; e < 4; ++e) { o[e] = f2bf(f0[e]); o[e + 4] = f2bf(f1[e]); }
    *(v8bf*)(x_bf + (size_t)i * 8) = o;
}

// ---- kernel 2: pack w_ih / w_hh into WMMA-B fragment-major bf16 -----------
// packed[((kt*NT + nt)*32 + lane)*16 + e] = w[nt*16 + lane%16][kt*32 + (e&7) + 8*(lane>=16) + 16*(e>=8)]

__global__ void pack_w_kernel(const float* __restrict__ wih,
                              const float* __restrict__ whh,
                              bf16_t* __restrict__ wih_pk,
                              bf16_t* __restrict__ whh_pk) {
    int o = blockIdx.x * blockDim.x + threadIdx.x;
    const int total = KT * NT * 32 * 16;               // 786432 = 1536*512
    if (o >= total) return;
    int e    = o & 15;
    int lane = (o >> 4) & 31;
    int nt   = (o >> 9) % NT;
    int kt   = o / (512 * NT);
    int g = nt * 16 + (lane & 15);
    int i = kt * 32 + (e & 7) + ((lane >> 4) << 3) + ((e >> 3) << 4);
    wih_pk[o] = f2bf(wih[(size_t)g * I_DIM + i]);
    whh_pk[o] = f2bf(whh[(size_t)g * H_DIM + i]);
}

// ---- kernel 3 (phase A): gx chunk GEMM ------------------------------------
// block = 128 threads (4 waves); wave tile = 64(M) x 64(N);
// grid = (MC/64, G3/256) = (64, 6). 16 WMMAs per 16 b128 loads per kt.

__global__ __launch_bounds__(128)
void gx_gemm_kernel(const bf16_t* __restrict__ x_bf,
                    const bf16_t* __restrict__ wih_pk,
                    const float* __restrict__ bih,
                    float* __restrict__ gx, int chunk) {
    const int lane  = threadIdx.x & 31;
    const int wave  = threadIdx.x >> 5;
    const int mbase = blockIdx.x * 64;                       // chunk-local row base
    const int nbase = blockIdx.y * 256 + wave * 64;          // gate-column base
    const int half8 = (lane < 16) ? 0 : 8;

    const bf16_t* row0 =
        x_bf + (size_t)(chunk * MC + mbase + (lane & 15)) * I_DIM;

    v8f acc[4][4];
#pragma unroll
    for (int mi = 0; mi < 4; ++mi)
#pragma unroll
        for (int nt = 0; nt < 4; ++nt) acc[mi][nt] = zero8();

#pragma unroll 2
    for (int kt = 0; kt < KT; ++kt) {
        int k0 = kt * 32 + half8;
        v16bf a[4];
#pragma unroll
        for (int mi = 0; mi < 4; ++mi)
            a[mi] = load_a_bf(row0 + (size_t)(mi * 16) * I_DIM, k0);
#pragma unroll
        for (int nt = 0; nt < 4; ++nt) {
            v16bf b = load_b_pk(wih_pk, kt, (nbase >> 4) + nt, lane);
#pragma unroll
            for (int mi = 0; mi < 4; ++mi)
                acc[mi][nt] = wmma_bf16(a[mi], b, acc[mi][nt]);
        }
    }

#pragma unroll
    for (int mi = 0; mi < 4; ++mi) {
        int mrow = mbase + mi * 16 + half8;
#pragma unroll
        for (int nt = 0; nt < 4; ++nt) {
            int g = nbase + nt * 16 + (lane & 15);
            float bias = bih[g];
#pragma unroll
            for (int v = 0; v < 8; ++v)
                gx[(size_t)(mrow + v) * G3 + g] = acc[mi][nt][v] + bias;
        }
    }
}

// ---- kernel 4 (phase B): persistent GRU scan over one time chunk ----------
// 32 WGs x 128 threads. WG bg owns hidden units [bg*16, bg*16+16).
// wave mt handles batch rows [mt*16, mt*16+16). Gates r/z/n for the same
// (batch, hidden) element land in the same lane slot -> branch-free update.

__global__ __launch_bounds__(128)
void gru_scan_kernel(const float* __restrict__ gx,
                     const bf16_t* __restrict__ whh_pk,
                     const float* __restrict__ bhh,
                     const int* __restrict__ lengths,
                     float* __restrict__ h_f32,
                     bf16_t* __restrict__ h_bf2,     // 2 ping-pong buffers
                     float* __restrict__ out,
                     int* __restrict__ bar, int chunk) {
    __shared__ __attribute__((aligned(16))) bf16_t lds_w[3 * KT * 512];  // 48 KB

    const int lane  = threadIdx.x & 31;
    const int mt    = threadIdx.x >> 5;          // 0..3, batch tile
    const int half8 = (lane < 16) ? 0 : 8;
    const int bg    = blockIdx.x;                // 0..31
    const int j     = (bg << 4) + (lane & 15);   // this lane's hidden column

    // Stage this WG's 3 gate-column tiles of packed w_hh into LDS (once).
    for (int d = threadIdx.x; d < 3 * KT * 256; d += blockDim.x) {  // dwords
        int gi = d / (KT * 256);
        int r  = d % (KT * 256);
        int kt = r / 256;
        int w  = r % 256;
        const unsigned* s =
            (const unsigned*)(whh_pk + (size_t)(kt * NT + gi * 32 + bg) * 512);
        ((unsigned*)lds_w)[(gi * KT + kt) * 256 + w] = s[w];
    }
    __syncthreads();

    const float bhh_r = bhh[j];
    const float bhh_z = bhh[H_DIM + j];
    const float bhh_n = bhh[2 * H_DIM + j];
    int lenv[8];
#pragma unroll
    for (int v = 0; v < 8; ++v) lenv[v] = lengths[mt * 16 + half8 + v];

    for (int tl = 0; tl < TC; ++tl) {
        const int t = chunk * TC + tl;

        // Prefetch next timestep's gx rows for this wave's slice.
        if (lane < 16 && tl + 1 < TC) {
            const float* p =
                gx + (size_t)((tl + 1) * B_DIM + mt * 16 + lane) * G3;
            __builtin_prefetch(p + (bg << 4), 0, 1);
            __builtin_prefetch(p + H_DIM + (bg << 4), 0, 1);
            __builtin_prefetch(p + 2 * H_DIM + (bg << 4), 0, 1);
        }

        const bf16_t* hsrc = h_bf2 + (size_t)(t & 1) * (B_DIM * H_DIM);
        bf16_t*       hdst = (bf16_t*)h_bf2 + (size_t)((t + 1) & 1) * (B_DIM * H_DIM);
        const bf16_t* hrow = hsrc + (size_t)(mt * 16 + (lane & 15)) * H_DIM;

        v8f cr = zero8(), cz = zero8(), cn = zero8();
#pragma unroll 4
        for (int kt = 0; kt < KT; ++kt) {
            v16bf a  = load_a_bf(hrow, kt * 32 + half8);
            v16bf br = load_b_lds(lds_w, 0, kt, lane);
            v16bf bz = load_b_lds(lds_w, 1, kt, lane);
            v16bf bn = load_b_lds(lds_w, 2, kt, lane);
            cr = wmma_bf16(a, br, cr);
            cz = wmma_bf16(a, bz, cz);
            cn = wmma_bf16(a, bn, cn);
        }

#pragma unroll
        for (int v = 0; v < 8; ++v) {
            int m = mt * 16 + half8 + v;
            size_t gxrow = (size_t)(tl * B_DIM + m) * G3;
            float xr = gx[gxrow + j];
            float xz = gx[gxrow + H_DIM + j];
            float xn = gx[gxrow + 2 * H_DIM + j];
            float hr = cr[v] + bhh_r;
            float hz = cz[v] + bhh_z;
            float hn = cn[v] + bhh_n;
            float r  = sigmoid_f(xr + hr);
            float z  = sigmoid_f(xz + hz);
            float nn = tanh_f(xn + r * hn);
            float hold = h_f32[(size_t)m * H_DIM + j];
            float hnew = (1.f - z) * nn + z * hold;
            bool  act   = t < lenv[v];
            float hnext = act ? hnew : hold;
            out[((size_t)t * B_DIM + m) * H_DIM + j] = act ? hnew : 0.f;
            h_f32[(size_t)m * H_DIM + j] = hnext;
            hdst[(size_t)m * H_DIM + j]  = f2bf(hnext);
            if (t == T_DIM - 1)  // h_last tail of d_out
                out[(size_t)T_DIM * B_DIM * H_DIM + (size_t)m * H_DIM + j] = hnext;
        }

        // Device-wide barrier across the 32 persistent workgroups.
        __threadfence();
        __syncthreads();
        if (threadIdx.x == 0) {
            if (__hip_atomic_fetch_add(&bar[0], 1, __ATOMIC_ACQ_REL,
                                       __HIP_MEMORY_SCOPE_AGENT) == NWG_B - 1) {
                __hip_atomic_store(&bar[0], 0, __ATOMIC_RELAXED,
                                   __HIP_MEMORY_SCOPE_AGENT);
                __hip_atomic_fetch_add(&bar[1], 1, __ATOMIC_ACQ_REL,
                                       __HIP_MEMORY_SCOPE_AGENT);
            }
            while (__hip_atomic_load(&bar[1], __ATOMIC_ACQUIRE,
                                     __HIP_MEMORY_SCOPE_AGENT) <= tl)
                __builtin_amdgcn_s_sleep(1);
        }
        __syncthreads();
    }
}

// ---- launcher -------------------------------------------------------------
// inputs: 0=x[T,B,I] f32, 1=lengths[B] i32, 2=w_ih[3H,I], 3=w_hh[3H,H],
//         4=b_ih[3H], 5=b_hh[3H]
// out: output[T,B,H] ++ h_last[1,B,H]  (f32)

extern "C" void kernel_launch(void* const* d_in, const int* in_sizes, int n_in,
                              void* d_out, int out_size, void* d_ws, size_t ws_size,
                              hipStream_t stream) {
    const float* x    = (const float*)d_in[0];
    const int*   len  = (const int*)d_in[1];
    const float* wih  = (const float*)d_in[2];
    const float* whh  = (const float*)d_in[3];
    const float* bih  = (const float*)d_in[4];
    const float* bhh  = (const float*)d_in[5];
    float* out = (float*)d_out;

    // workspace layout (bytes), all offsets 256B-aligned
    char* ws = (char*)d_ws;
    int*    bars   = (int*)ws;                                   // 1 KB
    float*  h_f32  = (float*)(ws + 1024);                        // 128 KB
    bf16_t* h_bf2  = (bf16_t*)(ws + 1024 + 131072);              // 128 KB (x2 bufs)
    bf16_t* wih_pk = (bf16_t*)(ws + 1024 + 131072 + 131072);     // 1.5 MB
    bf16_t* whh_pk = (bf16_t*)((char*)wih_pk + (size_t)KT * NT * 512 * 2);
    bf16_t* x_bf   = (bf16_t*)((char*)whh_pk + (size_t)KT * NT * 512 * 2);  // 64 MB
    float*  gx     = (float*)((char*)x_bf +
                              (size_t)T_DIM * B_DIM * I_DIM * 2);           // 25 MB

    init_state_kernel<<<512, 256, 0, stream>>>(h_f32, h_bf2, bars);
    pack_w_kernel<<<(KT * NT * 512 + 255) / 256, 256, 0, stream>>>(
        wih, whh, wih_pk, whh_pk);
    {
        int n8 = T_DIM * B_DIM * I_DIM / 8;     // 4,194,304
        cvt_x_kernel<<<(n8 + 255) / 256, 256, 0, stream>>>(x, x_bf, n8);
    }

    dim3 gridA(MC / 64, G3 / 256);   // (64, 6)
    for (int c = 0; c < NCHUNK; ++c) {
        gx_gemm_kernel<<<gridA, 128, 0, stream>>>(x_bf, wih_pk, bih, gx, c);
        gru_scan_kernel<<<NWG_B, 128, 0, stream>>>(
            gx, whh_pk, bhh, len, h_f32, h_bf2, out, bars + 2 * c, c);
    }
}
